// ModelIAS_59270548685358
// MI455X (gfx1250) — matrix operand order repaired and checked by reading, not compiled, not run on hardware
//
#include <hip/hip_runtime.h>

typedef __attribute__((ext_vector_type(16))) _Float16 v16h;
typedef __attribute__((ext_vector_type(8)))  _Float16 v8h;
typedef __attribute__((ext_vector_type(8)))  float    v8f;

#define B_   64
#define S_   512
#define H_   768
#define OUT_ 128
#define SM1  511   // S-1 output positions

// ---------------------------------------------------------------------------
// Kernel 1: segment-mean pooling. One wave32 per (batch, word).
// aligned[b, w, :] = mean of hidden[b, p, :] over the contiguous span with
// segment_ids[b, p] == w (ids sorted + compact), stored as f16.
// Zero-count words are written as exact zeros (reference semantics).
// ---------------------------------------------------------------------------
__global__ void seg_mean_kernel(const float* __restrict__ hidden,
                                const int*   __restrict__ seg,
                                _Float16*    __restrict__ aligned) {
  const int wave = (blockIdx.x * blockDim.x + threadIdx.x) >> 5;
  const int lane = threadIdx.x & 31;
  const int b = wave >> 9;      // wave / 512
  const int w = wave & 511;     // word id
  const int* segrow = seg + b * S_;

  // lower_bound(w) and lower_bound(w+1) over sorted row (uniform -> scalar)
  int lo;
  {
    int l = 0, r = S_;
    while (l < r) { int m = (l + r) >> 1; if (segrow[m] < w) l = m + 1; else r = m; }
    lo = l;
  }
  int hi;
  {
    int l = lo, r = S_;
    while (l < r) { int m = (l + r) >> 1; if (segrow[m] < w + 1) l = m + 1; else r = m; }
    hi = l;
  }
  const int count = hi - lo;
  _Float16* dst = aligned + (size_t)(b * S_ + w) * H_;

  if (count <= 0) {
    #pragma unroll
    for (int i = 0; i < H_ / 32; ++i) dst[lane + 32 * i] = (_Float16)0.0f;
    return;
  }

  float acc[H_ / 32];
  #pragma unroll
  for (int i = 0; i < H_ / 32; ++i) acc[i] = 0.0f;
  for (int p = lo; p < hi; ++p) {
    const float* src = hidden + (size_t)(b * S_ + p) * H_;
    #pragma unroll
    for (int i = 0; i < H_ / 32; ++i) acc[i] += src[lane + 32 * i];
  }
  const float inv = 1.0f / (float)count;
  #pragma unroll
  for (int i = 0; i < H_ / 32; ++i) dst[lane + 32 * i] = (_Float16)(acc[i] * inv);
}

// ---------------------------------------------------------------------------
// Kernel 2: W fp32 -> f16 (192 KB; resident in L2 for the whole GEMM)
// ---------------------------------------------------------------------------
__global__ void wcvt_kernel(const float* __restrict__ W,
                            _Float16*    __restrict__ Wh) {
  const int i = blockIdx.x * blockDim.x + threadIdx.x;
  if (i < OUT_ * H_) Wh[i] = (_Float16)W[i];
}

// ---------------------------------------------------------------------------
// WMMA fragment loaders (CDNA5 16-bit layouts, wave32)
// A 16x32: lane<16 -> M=lane,    K = {0..7, 16..23} (+k0)
//          lane>=16 -> M=lane-16, K = {8..15,24..31} (+k0)
// ---------------------------------------------------------------------------
__device__ __forceinline__ v16h load_fragA(const _Float16* __restrict__ row,
                                           int k0, int lane) {
  const int off = k0 + ((lane >> 4) << 3);          // +0 or +8
  const v8h lo = *(const v8h*)(row + off);          // K off..off+7
  const v8h hi = *(const v8h*)(row + off + 16);     // K off+16..off+23
  v16h r;
  #pragma unroll
  for (int i = 0; i < 8; ++i) { r[i] = lo[i]; r[i + 8] = hi[i]; }
  return r;
}

// B 32x16: lane<16 -> N=lane,    K = {0..7, 16..23} (+k0)
//          lane>=16 -> N=lane-16, K = {8..15,24..31} (+k0)
__device__ __forceinline__ v16h load_fragB(const _Float16* __restrict__ row,
                                           int k0, int lane) {
  const int off = k0 + ((lane >> 4) << 3);          // +0 or +8
  const v8h lo = *(const v8h*)(row + off);
  const v8h hi = *(const v8h*)(row + off + 16);
  v16h r;
  #pragma unroll
  for (int i = 0; i < 8; ++i) { r[i] = lo[i]; r[i + 8] = hi[i]; }
  return r;
}

// ---------------------------------------------------------------------------
// Kernel 3: GEMM. One wave per (batch, 16-row s-tile, OUT-half): 16x64 strip,
// 4 accumulators, K=768 in 24 steps of v_wmma_f32_16x16x32_f16.
// Explicit double-buffered software pipeline, FULLY UNROLLED: SSA renaming
// removes the loop-carried ping-pong copies, and every k-offset folds into
// the global_load_b128 immediate offset (no per-step address arithmetic).
// ---------------------------------------------------------------------------
__global__ void gemm_kernel(const _Float16* __restrict__ aligned,
                            const _Float16* __restrict__ Wh,
                            const float*    __restrict__ bias,
                            float*          __restrict__ out) {
  const int wave = (blockIdx.x * blockDim.x + threadIdx.x) >> 5;
  const int lane = threadIdx.x & 31;
  const int b   = wave >> 6;       // 64 waves per batch
  const int rem = wave & 63;
  const int mt  = rem >> 1;        // 32 m-tiles
  const int nh  = rem & 1;         // OUT half: n-tiles nh*4 .. nh*4+3
  const int m0  = mt * 16;         // s-tile base
  const int lrow = lane & 15;

  // A row: word = s + 1; clamp the single OOB row (word 512) — its D row
  // (s = 511) is never stored.
  int word = m0 + lrow + 1;
  if (word > S_ - 1) word = S_ - 1;
  const _Float16* arow = aligned + (size_t)(b * S_ + word) * H_;

  // B row pointers for this wave's 4 OUT tiles
  const _Float16* brow[4];
  #pragma unroll
  for (int n = 0; n < 4; ++n)
    brow[n] = Wh + (size_t)((nh * 4 + n) * 16 + lrow) * H_;

  v8f acc[4] = {};

  // ---- software pipeline: prologue loads chunk 0 into buffer 0 ----
  v16h a0 = load_fragA(arow, 0, lane);
  v16h b0[4], b1[4];
  #pragma unroll
  for (int n = 0; n < 4; ++n) b0[n] = load_fragB(brow[n], 0, lane);
  v16h a1;

  // Steady state (fully unrolled): consume chunks kc (buf0) and kc+1 (buf1),
  // prefetch kc+1 into buf1 and kc+2 into buf0. Chunks 0..21.
  #pragma unroll
  for (int kc = 0; kc < 22; kc += 2) {
    const int k1 = (kc + 1) * 32;
    const int k2 = (kc + 2) * 32;

    a1 = load_fragA(arow, k1, lane);
    #pragma unroll
    for (int n = 0; n < 4; ++n) b1[n] = load_fragB(brow[n], k1, lane);

    #pragma unroll
    for (int n = 0; n < 4; ++n)
      acc[n] = __builtin_amdgcn_wmma_f32_16x16x32_f16(
          false, a0, false, b0[n], (short)0, acc[n], false, false);

    a0 = load_fragA(arow, k2, lane);
    #pragma unroll
    for (int n = 0; n < 4; ++n) b0[n] = load_fragB(brow[n], k2, lane);

    #pragma unroll
    for (int n = 0; n < 4; ++n)
      acc[n] = __builtin_amdgcn_wmma_f32_16x16x32_f16(
          false, a1, false, b1[n], (short)0, acc[n], false, false);
  }

  // ---- epilogue: chunk 22 already in buf0; load chunk 23, then finish ----
  a1 = load_fragA(arow, 23 * 32, lane);
  #pragma unroll
  for (int n = 0; n < 4; ++n) b1[n] = load_fragB(brow[n], 23 * 32, lane);
  #pragma unroll
  for (int n = 0; n < 4; ++n)
    acc[n] = __builtin_amdgcn_wmma_f32_16x16x32_f16(
        false, a0, false, b0[n], (short)0, acc[n], false, false);
  #pragma unroll
  for (int n = 0; n < 4; ++n)
    acc[n] = __builtin_amdgcn_wmma_f32_16x16x32_f16(
        false, a1, false, b1[n], (short)0, acc[n], false, false);

  // D layout: VGPR j, lanes 0-15: (M=j, N=lane); lanes 16-31: (M=j+8, N=lane-16)
  const int half = lane >> 4;
  #pragma unroll
  for (int n = 0; n < 4; ++n) {
    const int o = (nh * 4 + n) * 16 + lrow;
    const float bv = bias[o];
    float* obase = out + (size_t)(b * OUT_ + o) * SM1;
    #pragma unroll
    for (int j = 0; j < 8; ++j) {
      const int s = m0 + j + half * 8;
      if (s < SM1) obase[s] = acc[n][j] + bv;   // 8 consecutive floats per lane
    }
  }
}

// ---------------------------------------------------------------------------
extern "C" void kernel_launch(void* const* d_in, const int* in_sizes, int n_in,
                              void* d_out, int out_size, void* d_ws, size_t ws_size,
                              hipStream_t stream) {
  (void)in_sizes; (void)n_in; (void)out_size; (void)ws_size;
  const float* hidden = (const float*)d_in[0];   // [64,512,768] f32
  const float* W      = (const float*)d_in[1];   // [128,768]    f32
  const float* bias   = (const float*)d_in[2];   // [128]        f32
  const int*   seg    = (const int*)  d_in[3];   // [64,512]     i32
  float*       out    = (float*)d_out;           // [64,128,511] f32

  _Float16* alignedh = (_Float16*)d_ws;                           // 50,331,648 B
  _Float16* Wh = (_Float16*)((char*)d_ws +
                             (size_t)B_ * S_ * H_ * sizeof(_Float16));

  // 64*512 words, one wave each: 32768 waves / 8 per block = 4096 blocks
  seg_mean_kernel<<<dim3(4096), dim3(256), 0, stream>>>(hidden, seg, alignedh);
  wcvt_kernel<<<dim3((OUT_ * H_ + 255) / 256), dim3(256), 0, stream>>>(W, Wh);
  // 64 batches * 32 m-tiles * 2 OUT-halves = 4096 waves / 8 per block = 512 blocks
  gemm_kernel<<<dim3(512), dim3(256), 0, stream>>>(alignedh, Wh, bias, out);
}
